// DatasetGraphGCN_60739427500571
// MI455X (gfx1250) — compile-verified, matrix operand order checked by reading, not run on hardware
//
#include <hip/hip_runtime.h>
#include <hip/hip_bf16.h>

// ---------------------------------------------------------------------------
// Types for CDNA5 WMMA (wave32): v_wmma_f32_16x16x32_bf16
// ---------------------------------------------------------------------------
typedef __attribute__((ext_vector_type(16))) __bf16     v16bf;
typedef __attribute__((ext_vector_type(8)))  float      v8f;
typedef __attribute__((ext_vector_type(4)))  unsigned int u32x4;

struct Frag32B { u32x4 lo, hi; };   // 32 bytes == v16bf

__device__ __forceinline__ unsigned short f32_to_bf16_rne(float f) {
  unsigned int u = __float_as_uint(f);
  u += 0x7fffu + ((u >> 16) & 1u);          // round-to-nearest-even
  return (unsigned short)(u >> 16);
}

// ---------------------------------------------------------------------------
// Elementwise helpers
// ---------------------------------------------------------------------------
__global__ void k_f2bf(const float* __restrict__ in, unsigned short* __restrict__ out, int n) {
  int i = blockIdx.x * blockDim.x + threadIdx.x;
  if (i < n) out[i] = f32_to_bf16_rne(in[i]);
}

// W[K][N] f32 row-major  ->  Wt[N][K] bf16 row-major (B operand wants per-lane
// contiguous K, i.e. column-of-B == row-of-Bt).
__global__ void k_transpose_f2bf(const float* __restrict__ W, unsigned short* __restrict__ Wt,
                                 int K, int N) {
  int i = blockIdx.x * blockDim.x + threadIdx.x;
  if (i >= K * N) return;
  int k = i / N, n = i - k * N;
  Wt[(size_t)n * K + k] = f32_to_bf16_rne(W[i]);
}

__global__ void k_fill1(float* __restrict__ p, int n) {
  int i = blockIdx.x * blockDim.x + threadIdx.x;
  if (i < n) p[i] = 1.0f;                    // self-loop weight contributes 1.0 to deg
}

__global__ void k_deg_edges(const int* __restrict__ col, const float* __restrict__ ew,
                            float* __restrict__ deg, int E) {
  int e = blockIdx.x * blockDim.x + threadIdx.x;
  if (e < E) atomicAdd(&deg[col[e]], ew[e]);
}

__global__ void k_rsqrt_inplace(float* __restrict__ p, int n) {
  int i = blockIdx.x * blockDim.x + threadIdx.x;
  if (i < n) p[i] = rsqrtf(p[i]);            // deg >= 1.0 always (self loops)
}

// acc already holds the residual GEMM result; fold in self-loop message
// (norm = dis^2), conv bias b, residual bias rb.
__global__ void k_init_acc(float* __restrict__ acc, const float* __restrict__ xw,
                           const float* __restrict__ b, const float* __restrict__ rb,
                           const float* __restrict__ dis, int n, int F) {
  int i = blockIdx.x * blockDim.x + threadIdx.x;
  if (i >= n * F) return;
  int node = i / F, f = i - node * F;
  float d = dis[node];
  acc[i] = acc[i] + xw[i] * (d * d) + b[f] + rb[f];
}

// One block per edge: acc[dst] += xw[src] * dis[src]*ew*dis[dst]
__global__ void k_scatter(const float* __restrict__ xw, const int* __restrict__ row,
                          const int* __restrict__ col, const float* __restrict__ ew,
                          const float* __restrict__ dis, float* __restrict__ acc,
                          int E, int F) {
  int e = blockIdx.x;
  if (e >= E) return;
  int src = row[e], dst = col[e];
  float coef = dis[src] * ew[e] * dis[dst];
  const float* xs = xw + (size_t)src * F;
  float*       ad = acc + (size_t)dst * F;
  for (int f = threadIdx.x; f < F; f += blockDim.x)
    atomicAdd(&ad[f], xs[f] * coef);
}

__global__ void k_relu_f2bf(const float* __restrict__ in, unsigned short* __restrict__ out, int n) {
  int i = blockIdx.x * blockDim.x + threadIdx.x;
  if (i < n) out[i] = f32_to_bf16_rne(fmaxf(in[i], 0.0f));
}

// ---------------------------------------------------------------------------
// bf16 WMMA GEMM: C[M][N] = A[M][K] @ B[K][N], with B given transposed as
// Bt[N][K]. Each wave owns a 32x64 C tile (2 M-frags x 4 N-frags).
// Fragment layout (16-bit A 16x32, lane L: l=L&15 row, hi=L>>4):
//   elems 0..7  = row[k0 + hi*8 .. +7]
//   elems 8..15 = row[k0 + 16 + hi*8 .. +7]
// Both halves are aligned 16B loads (global_load_b128). B mirrors A on Bt rows.
// ---------------------------------------------------------------------------
__device__ __forceinline__ v16bf load_frag(const unsigned short* __restrict__ rowp,
                                           int k0, int hi) {
  Frag32B f;
  f.lo = *(const u32x4*)(rowp + k0 + hi * 8);
  f.hi = *(const u32x4*)(rowp + k0 + 16 + hi * 8);
  return __builtin_bit_cast(v16bf, f);
}

#define WMMA_BF16(A_, B_, C_) \
  __builtin_amdgcn_wmma_f32_16x16x32_bf16(false, (A_), false, (B_), (short)0, (C_), false, false)

__global__ void __launch_bounds__(256)
k_gemm_bf16_wmma(const unsigned short* __restrict__ A,   // [M][K] bf16 row-major
                 const unsigned short* __restrict__ Bt,  // [N][K] bf16 row-major
                 float* __restrict__ C,                   // [M][N] f32
                 int M, int N, int K) {
  const int wave   = blockIdx.x * (blockDim.x >> 5) + (threadIdx.x >> 5);
  const int lane   = threadIdx.x & 31;
  const int nTiles = N >> 6;                 // 64-wide wave tiles along N
  const int mTile  = wave / nTiles;
  const int nTile  = wave - mTile * nTiles;
  const int mTiles = (M + 31) >> 5;
  if (mTile >= mTiles) return;               // wave-uniform exit (EXEC all-1 inside)

  const int l = lane & 15, hi = lane >> 4;
  const int m0 = mTile * 32;
  const int n0 = nTile * 64;

  int ra0 = m0 + l;       if (ra0 > M - 1) ra0 = M - 1;   // clamp (data-only, EXEC unchanged)
  int ra1 = m0 + 16 + l;  if (ra1 > M - 1) ra1 = M - 1;
  const unsigned short* arow0 = A + (size_t)ra0 * K;
  const unsigned short* arow1 = A + (size_t)ra1 * K;
  const unsigned short* brow0 = Bt + (size_t)(n0      + l) * K;
  const unsigned short* brow1 = Bt + (size_t)(n0 + 16 + l) * K;
  const unsigned short* brow2 = Bt + (size_t)(n0 + 32 + l) * K;
  const unsigned short* brow3 = Bt + (size_t)(n0 + 48 + l) * K;

  v8f acc00 = {}, acc01 = {}, acc02 = {}, acc03 = {};
  v8f acc10 = {}, acc11 = {}, acc12 = {}, acc13 = {};

  #pragma unroll 4
  for (int k0 = 0; k0 < K; k0 += 32) {
    v16bf a0 = load_frag(arow0, k0, hi);
    v16bf a1 = load_frag(arow1, k0, hi);
    v16bf b0 = load_frag(brow0, k0, hi);
    v16bf b1 = load_frag(brow1, k0, hi);
    v16bf b2 = load_frag(brow2, k0, hi);
    v16bf b3 = load_frag(brow3, k0, hi);
    acc00 = WMMA_BF16(a0, b0, acc00);
    acc01 = WMMA_BF16(a0, b1, acc01);
    acc02 = WMMA_BF16(a0, b2, acc02);
    acc03 = WMMA_BF16(a0, b3, acc03);
    acc10 = WMMA_BF16(a1, b0, acc10);
    acc11 = WMMA_BF16(a1, b1, acc11);
    acc12 = WMMA_BF16(a1, b2, acc12);
    acc13 = WMMA_BF16(a1, b3, acc13);
  }

  // C/D layout: lane column = l, rows = hi*8 + r (second frag +16)
  const int colb = n0 + l;
  #pragma unroll
  for (int r = 0; r < 8; ++r) {
    int row0 = m0 + hi * 8 + r;
    int row1 = row0 + 16;
    if (row0 < M) {
      float* c = C + (size_t)row0 * N + colb;
      c[0] = acc00[r]; c[16] = acc01[r]; c[32] = acc02[r]; c[48] = acc03[r];
    }
    if (row1 < M) {
      float* c = C + (size_t)row1 * N + colb;
      c[0] = acc10[r]; c[16] = acc11[r]; c[32] = acc12[r]; c[48] = acc13[r];
    }
  }
}

// ---------------------------------------------------------------------------
// Host-side orchestration
// ---------------------------------------------------------------------------
static inline int cdiv(long a, long b) { return (int)((a + b - 1) / b); }

extern "C" void kernel_launch(void* const* d_in, const int* in_sizes, int n_in,
                              void* d_out, int out_size, void* d_ws, size_t ws_size,
                              hipStream_t stream) {
  // setup_inputs order: x, w1, b1, w2, b2, rw1, rb1, rw2, rb2, edge_weight, edge_index
  const float* x   = (const float*)d_in[0];
  const float* w1  = (const float*)d_in[1];
  const float* b1  = (const float*)d_in[2];
  const float* w2  = (const float*)d_in[3];
  const float* b2  = (const float*)d_in[4];
  const float* rw1 = (const float*)d_in[5];
  const float* rb1 = (const float*)d_in[6];
  const float* rw2 = (const float*)d_in[7];
  const float* rb2 = (const float*)d_in[8];
  const float* ew  = (const float*)d_in[9];
  const int*   ei  = (const int*)  d_in[10];

  const int H    = in_sizes[2];                 // 1024
  const int OUT  = in_sizes[4];                 // 128
  const int F_IN = in_sizes[1] / H;             // 512
  const int Nn   = in_sizes[0] / F_IN;          // 20000
  const int E    = in_sizes[9];                 // 200000
  const int* row = ei;                          // edge_index[0] = sources
  const int* col = ei + E;                      // edge_index[1] = targets

  // ---- workspace carving (256B aligned) ----
  char* ws = (char*)d_ws;
  size_t off = 0;
  auto carve = [&](size_t bytes) { size_t r = off; off = (off + bytes + 255) & ~(size_t)255; return r; };
  float*          dis   = (float*)         (ws + carve((size_t)Nn * 4));
  unsigned short* w1t   = (unsigned short*)(ws + carve((size_t)F_IN * H * 2));
  unsigned short* rw1t  = (unsigned short*)(ws + carve((size_t)F_IN * H * 2));
  unsigned short* w2t   = (unsigned short*)(ws + carve((size_t)H * OUT * 2));
  unsigned short* rw2t  = (unsigned short*)(ws + carve((size_t)H * OUT * 2));
  unsigned short* xbf   = (unsigned short*)(ws + carve((size_t)Nn * F_IN * 2));
  size_t o_xw1          = carve((size_t)Nn * H * 4);
  float*          xw1   = (float*)(ws + o_xw1);
  float*          res1  = (float*)(ws + carve((size_t)Nn * H * 4));   // becomes acc1
  // After layer-1 scatter, the xw1 region is dead: alias h_bf16 + xw2 into it.
  unsigned short* hbf   = (unsigned short*)(ws + o_xw1);
  size_t hbf_bytes      = (((size_t)Nn * H * 2) + 255) & ~(size_t)255;
  float*          xw2   = (float*)(ws + o_xw1 + hbf_bytes);
  float*          out   = (float*)d_out;

  const int T = 256;

  // ---- precision conversion & weight transposes ----
  k_f2bf<<<cdiv((long)Nn * F_IN, T), T, 0, stream>>>(x, xbf, Nn * F_IN);
  k_transpose_f2bf<<<cdiv((long)F_IN * H, T), T, 0, stream>>>(w1,  w1t,  F_IN, H);
  k_transpose_f2bf<<<cdiv((long)F_IN * H, T), T, 0, stream>>>(rw1, rw1t, F_IN, H);
  k_transpose_f2bf<<<cdiv((long)H * OUT, T), T, 0, stream>>>(w2,  w2t,  H, OUT);
  k_transpose_f2bf<<<cdiv((long)H * OUT, T), T, 0, stream>>>(rw2, rw2t, H, OUT);

  // ---- symmetric normalization: dis = rsqrt(1 + sum_in ew) ----
  k_fill1<<<cdiv(Nn, T), T, 0, stream>>>(dis, Nn);
  k_deg_edges<<<cdiv(E, T), T, 0, stream>>>(col, ew, dis, E);
  k_rsqrt_inplace<<<cdiv(Nn, T), T, 0, stream>>>(dis, Nn);

  // ---- layer 1: xw1 = x@w1, res1 = x@rw1 (WMMA bf16) ----
  {
    int waves = ((Nn + 31) / 32) * (H / 64);
    int blocks = (waves + 7) / 8;
    k_gemm_bf16_wmma<<<blocks, 256, 0, stream>>>(xbf, w1t,  xw1,  Nn, H, F_IN);
    k_gemm_bf16_wmma<<<blocks, 256, 0, stream>>>(xbf, rw1t, res1, Nn, H, F_IN);
  }
  k_init_acc<<<cdiv((long)Nn * H, T), T, 0, stream>>>(res1, xw1, b1, rb1, dis, Nn, H);
  k_scatter<<<E, 256, 0, stream>>>(xw1, row, col, ew, dis, res1, E, H);
  k_relu_f2bf<<<cdiv((long)Nn * H, T), T, 0, stream>>>(res1, hbf, Nn * H);

  // ---- layer 2: xw2 = h@w2, out = h@rw2 (+ aggregation into out) ----
  {
    int waves = ((Nn + 31) / 32) * (OUT / 64);
    int blocks = (waves + 7) / 8;
    k_gemm_bf16_wmma<<<blocks, 256, 0, stream>>>(hbf, w2t,  xw2, Nn, OUT, H);
    k_gemm_bf16_wmma<<<blocks, 256, 0, stream>>>(hbf, rw2t, out, Nn, OUT, H);
  }
  k_init_acc<<<cdiv((long)Nn * OUT, T), T, 0, stream>>>(out, xw2, b2, rb2, dis, Nn, OUT);
  k_scatter<<<E, 128, 0, stream>>>(xw2, row, col, ew, dis, out, E, OUT);
}
